// EHGN_TopK_Sparse_51453708206092
// MI455X (gfx1250) — compile-verified, matrix operand order checked by reading, not run on hardware
//
#include <hip/hip_runtime.h>
#include <hip/hip_bf16.h>

// ---------------------------------------------------------------------------
// Problem constants (from the reference)
// ---------------------------------------------------------------------------
#define NGRAPHS 20
#define NPER    1000
#define NNODES  20000
#define NEDGES  320000
#define TOPK    5
#define NPOOL   (NGRAPHS * TOPK)     // 100
#define DIM     128                  // NODE_DIM == MSG_DIM == HID
#define EDIM    16
#define E1K     273                  // 2*128 + 1 + 16
#define E1KT    9                    // ceil(273/32)
#define NT8     8                    // 128 / 16 output tiles

#define EDGE_TILES (NEDGES / 16)     // 20000
#define EDGE_BLOCKS 512
#define MAX_GEMM_BLOCKS 256

typedef __attribute__((ext_vector_type(16))) __bf16 v16bf;
typedef __attribute__((ext_vector_type(8)))  __bf16 v8bf;
typedef __attribute__((ext_vector_type(8)))  float  v8f;
typedef int v4i_ __attribute__((__vector_size__(16)));

#define HAS_ASYNC_LDS __has_builtin(__builtin_amdgcn_global_load_async_to_lds_b128)

__device__ __forceinline__ v16bf concat8(v8bf lo, v8bf hi) {
  return __builtin_shufflevector(lo, hi, 0,1,2,3,4,5,6,7,8,9,10,11,12,13,14,15);
}

__device__ __forceinline__ float silu_f(float x) { return x / (1.0f + __expf(-x)); }

// Stage a block of packed weights from global into LDS.  Prefer the CDNA5
// async copy path (GLOBAL_LOAD_ASYNC_TO_LDS_B128, ASYNCcnt); fall back to a
// plain vectorized copy if the builtin is unavailable on this toolchain.
__device__ __forceinline__ void stage_lds(const __bf16* __restrict__ g,
                                          __bf16* l, int nbytes) {
#if HAS_ASYNC_LDS
  typedef __attribute__((address_space(1))) v4i_ gv4;
  typedef __attribute__((address_space(3))) v4i_ lv4;
  gv4* gp = (gv4*)g;   // C-style cast: drops const, generic->global AS
  lv4* lp = (lv4*)l;   // generic->LDS AS
  int n16 = nbytes >> 4;
  for (int o = threadIdx.x; o < n16; o += 128)
    __builtin_amdgcn_global_load_async_to_lds_b128(gp + o, lp + o, 0, 0);
#else
  const uint4* gs = (const uint4*)g;
  uint4* ls = (uint4*)l;
  int n16 = nbytes >> 4;
  for (int o = threadIdx.x; o < n16; o += 128) ls[o] = gs[o];
#endif
}

__device__ __forceinline__ void stage_fence() {
#if HAS_ASYNC_LDS
#if __has_builtin(__builtin_amdgcn_s_wait_asynccnt)
  __builtin_amdgcn_s_wait_asynccnt(0);
#else
  asm volatile("s_wait_asynccnt 0" ::: "memory");
#endif
#endif
  __syncthreads();
}

// A-fragment for v_wmma_f32_16x16x32_bf16 from an LDS tile (row-major, `stride`
// bf16 per row).  Lane L: row = L%16; lanes 0-15 hold K = k0+{0..7,16..23},
// lanes 16-31 hold K = k0+{8..15,24..31}  (ISA 7.12.2, 16-bit A 16x32).
__device__ __forceinline__ v16bf a_frag(const __bf16* t, int stride, int k0) {
  const int lane = threadIdx.x & 31;
  const int r = lane & 15, hf = lane >> 4;
  const v8bf* p0 = (const v8bf*)(t + r * stride + k0 + hf * 8);
  const v8bf* p1 = (const v8bf*)(t + r * stride + k0 + 16 + hf * 8);
  return concat8(*p0, *p1);
}

// B-fragment from a fragment-major pack (LDS-resident): fragment `fragIdx`
// is 32 lanes x 16 bf16 contiguous (1 KB), lane-major.
__device__ __forceinline__ v16bf b_frag(const __bf16* p, int fragIdx) {
  const int lane = threadIdx.x & 31;
  const v8bf* q = (const v8bf*)(p + ((size_t)fragIdx * 32 + lane) * 16);
  return concat8(q[0], q[1]);
}

__device__ __forceinline__ v8f wmma_bf16(v16bf a, v16bf b, v8f c) {
  return __builtin_amdgcn_wmma_f32_16x16x32_bf16(false, a, false, b, (short)0, c,
                                                 false, false);
}

// ---------------------------------------------------------------------------
// Weight packing: W[K,N] f32 -> bf16 fragment-major [nt][kt][lane][16]
// ---------------------------------------------------------------------------
__global__ void pack_w_kernel(const float* __restrict__ W, __bf16* __restrict__ out,
                              int K, int N, int KT, int NT) {
  int t = blockIdx.x * blockDim.x + threadIdx.x;
  int total = NT * KT * 512;
  if (t >= total) return;
  int j = t & 15;
  int lane = (t >> 4) & 31;
  int rest = t >> 9;
  int kt = rest % KT;
  int nt = rest / KT;
  int hf = lane >> 4;
  int k = kt * 32 + ((j < 8) ? (hf * 8 + j) : (16 + hf * 8 + (j - 8)));
  int n = nt * 16 + (lane & 15);
  float v = (k < K && n < N) ? W[(size_t)k * N + n] : 0.0f;
  out[t] = (__bf16)v;
}

// ---------------------------------------------------------------------------
// EGNN edge kernel (persistent, grid-stride over 16-edge tiles).
// Weights staged once per block into LDS; per wave:
//   A = [h[dst] | h[src] | d2 | edge_attr | pad]  (16 x 288 bf16, LDS)
//   e1: 9x8 WMMA, silu -> LDS; e2: 4x8 WMMA, silu
//   cw = xw(m) shuffle-reduction; atomics into magg / xagg (segment_sum).
// If new_id != nullptr (pooled stage): remap node ids + edge mask.
// ---------------------------------------------------------------------------
__global__ void __launch_bounds__(128) edge_kernel(
    const int* __restrict__ esrc, const int* __restrict__ edst,
    const float* __restrict__ eattr,
    const float* __restrict__ xn, const float* __restrict__ hn,
    const __bf16* __restrict__ w1p, const float* __restrict__ b1,
    const __bf16* __restrict__ w2p, const float* __restrict__ b2,
    const float* __restrict__ xwv, const float* __restrict__ xwb,
    const int* __restrict__ new_id,
    float* __restrict__ magg, float* __restrict__ xagg) {
  __shared__ alignas(16) __bf16 w1s[E1KT * NT8 * 512];   // 73728 B
  __shared__ alignas(16) __bf16 w2s[4 * NT8 * 512];      // 32768 B
  __shared__ alignas(16) __bf16 sh_a[4][16 * 288];       // 36864 B
  __shared__ float sh_d2[4][16];
  __shared__ float sh_rel[4][16][3];
  __shared__ float sh_msk[4][16];
  __shared__ int   sh_s[4][16];
  __shared__ int   sh_d[4][16];
  __shared__ float sh_cw[4][16];

  stage_lds(w1p, w1s, E1KT * NT8 * 512 * 2);
  stage_lds(w2p, w2s, 4 * NT8 * 512 * 2);
  stage_fence();

  const int lane = threadIdx.x & 31;
  const int wv = threadIdx.x >> 5;
  const int hf = lane >> 4, cl = lane & 15;
  const v8f vz = {0.f, 0.f, 0.f, 0.f, 0.f, 0.f, 0.f, 0.f};

  for (int tile = blockIdx.x * 4 + wv; tile < EDGE_TILES; tile += gridDim.x * 4) {
    const int ebase = tile * 16;

    if (lane < 16) {
      int e = ebase + lane;
      int s = esrc[e], d = edst[e];
      float mk = 1.0f;
      if (new_id) {
        int ns = new_id[s], nd = new_id[d];
        mk = (ns >= 0 && nd >= 0) ? 1.0f : 0.0f;
        s = ns < 0 ? 0 : ns;
        d = nd < 0 ? 0 : nd;
      }
      float r0 = xn[s * 3 + 0] - xn[d * 3 + 0];
      float r1 = xn[s * 3 + 1] - xn[d * 3 + 1];
      float r2 = xn[s * 3 + 2] - xn[d * 3 + 2];
      sh_rel[wv][lane][0] = r0;
      sh_rel[wv][lane][1] = r1;
      sh_rel[wv][lane][2] = r2;
      sh_d2[wv][lane] = r0 * r0 + r1 * r1 + r2 * r2;
      sh_msk[wv][lane] = mk;
      sh_s[wv][lane] = s;
      sh_d[wv][lane] = d;
    }

    // -- build A tile (coalesced row-by-row gather, f32 -> bf16) ------------
    for (int r = 0; r < 16; ++r) {
      int hs = sh_s[wv][r], hd = sh_d[wv][r];
      const float* hrow_d = hn + (size_t)hd * DIM;
      const float* hrow_s = hn + (size_t)hs * DIM;
#pragma unroll
      for (int j = 0; j < 9; ++j) {
        int col = lane + 32 * j;
        float v;
        if (col < 128)       v = hrow_d[col];
        else if (col < 256)  v = hrow_s[col - 128];
        else if (col == 256) v = sh_d2[wv][r];
        else if (col < 273)  v = eattr[(size_t)(ebase + r) * EDIM + (col - 257)];
        else                 v = 0.0f;
        sh_a[wv][r * 288 + col] = (__bf16)v;
      }
    }

    // -- e1: [16x288] @ [288x128] -------------------------------------------
    v8f acc[NT8];
#pragma unroll
    for (int nt = 0; nt < NT8; ++nt) acc[nt] = vz;
#pragma unroll
    for (int kt = 0; kt < E1KT; ++kt) {
      v16bf af = a_frag(sh_a[wv], 288, kt * 32);
#pragma unroll
      for (int nt = 0; nt < NT8; ++nt) {
        v16bf bf = b_frag(w1s, nt * E1KT + kt);
        acc[nt] = wmma_bf16(af, bf, acc[nt]);
      }
    }
    // bias + silu, C-layout -> A-layout through LDS (reuse sh_a cols 0..127)
#pragma unroll
    for (int nt = 0; nt < NT8; ++nt) {
      float bv = b1[nt * 16 + cl];
#pragma unroll
      for (int i = 0; i < 8; ++i) {
        float v = silu_f(acc[nt][i] + bv);
        sh_a[wv][(i + 8 * hf) * 288 + nt * 16 + cl] = (__bf16)v;
      }
    }

    // -- e2: [16x128] @ [128x128] -------------------------------------------
    v8f acc2[NT8];
#pragma unroll
    for (int nt = 0; nt < NT8; ++nt) acc2[nt] = vz;
#pragma unroll
    for (int kt = 0; kt < 4; ++kt) {
      v16bf af = a_frag(sh_a[wv], 288, kt * 32);
#pragma unroll
      for (int nt = 0; nt < NT8; ++nt) {
        v16bf bf = b_frag(w2s, nt * 4 + kt);
        acc2[nt] = wmma_bf16(af, bf, acc2[nt]);
      }
    }

    // -- bias + silu, xw partials, message atomics --------------------------
    float part[8];
#pragma unroll
    for (int i = 0; i < 8; ++i) part[i] = 0.0f;
    float xwl[NT8];
#pragma unroll
    for (int nt = 0; nt < NT8; ++nt) xwl[nt] = xwv[nt * 16 + cl];
    float mrow[8];
    int drow[8];
#pragma unroll
    for (int i = 0; i < 8; ++i) {
      mrow[i] = sh_msk[wv][i + 8 * hf];
      drow[i] = sh_d[wv][i + 8 * hf];
    }
#pragma unroll
    for (int nt = 0; nt < NT8; ++nt) {
      float bv = b2[nt * 16 + cl];
#pragma unroll
      for (int i = 0; i < 8; ++i) {
        float vpre = silu_f(acc2[nt][i] + bv);        // pre-mask message
        part[i] += vpre * xwl[nt];                    // xw uses pre-mask m
        atomicAdd(&magg[(size_t)drow[i] * DIM + nt * 16 + cl], vpre * mrow[i]);
      }
    }
    // reduce xw partials across the 16 lanes of each half (rows i + 8*hf)
#pragma unroll
    for (int i = 0; i < 8; ++i) {
      float p = part[i];
      p += __shfl_xor(p, 1, 32);
      p += __shfl_xor(p, 2, 32);
      p += __shfl_xor(p, 4, 32);
      p += __shfl_xor(p, 8, 32);
      float cwv = (p + xwb[0]) * mrow[i];
      if (cl == 0) sh_cw[wv][i + 8 * hf] = cwv;
    }
    // x aggregation: xagg[dst] += rel * cw
    if (lane < 16) {
      float cwm = sh_cw[wv][lane];
      int d = sh_d[wv][lane];
#pragma unroll
      for (int v = 0; v < 3; ++v)
        atomicAdd(&xagg[(size_t)d * 3 + v], sh_rel[wv][lane][v] * cwm);
    }
  }
}

// ---------------------------------------------------------------------------
// EGNN node kernel: h += h2(silu(h1([h | agg]))) ; x += xagg   (persistent)
// ---------------------------------------------------------------------------
__global__ void __launch_bounds__(128) node_kernel(
    float* __restrict__ xn, float* __restrict__ hn,
    const float* __restrict__ magg, const float* __restrict__ xagg,
    const __bf16* __restrict__ h1p, const float* __restrict__ hb1,
    const __bf16* __restrict__ h2p, const float* __restrict__ hb2, int n) {
  __shared__ alignas(16) __bf16 w1s[8 * NT8 * 512];   // 65536 B
  __shared__ alignas(16) __bf16 w2s[4 * NT8 * 512];   // 32768 B
  __shared__ alignas(16) __bf16 sh_a[4][16 * 256];    // 32768 B

  stage_lds(h1p, w1s, 8 * NT8 * 512 * 2);
  stage_lds(h2p, w2s, 4 * NT8 * 512 * 2);
  stage_fence();

  const int lane = threadIdx.x & 31;
  const int wv = threadIdx.x >> 5;
  const int hf = lane >> 4, cl = lane & 15;
  const v8f vz = {0.f, 0.f, 0.f, 0.f, 0.f, 0.f, 0.f, 0.f};
  const int ntiles = (n + 15) / 16;

  for (int tile = blockIdx.x * 4 + wv; tile < ntiles; tile += gridDim.x * 4) {
    const int nbase = tile * 16;

    for (int r = 0; r < 16; ++r) {
      int node = nbase + r;
      if (node > n - 1) node = n - 1;
#pragma unroll
      for (int j = 0; j < 8; ++j) {
        int col = lane + 32 * j;
        float v = (col < 128) ? hn[(size_t)node * DIM + col]
                              : magg[(size_t)node * DIM + (col - 128)];
        sh_a[wv][r * 256 + col] = (__bf16)v;
      }
    }

    v8f acc[NT8];
#pragma unroll
    for (int nt = 0; nt < NT8; ++nt) acc[nt] = vz;
#pragma unroll
    for (int kt = 0; kt < 8; ++kt) {
      v16bf af = a_frag(sh_a[wv], 256, kt * 32);
#pragma unroll
      for (int nt = 0; nt < NT8; ++nt) {
        v16bf bf = b_frag(w1s, nt * 8 + kt);
        acc[nt] = wmma_bf16(af, bf, acc[nt]);
      }
    }
#pragma unroll
    for (int nt = 0; nt < NT8; ++nt) {
      float bv = hb1[nt * 16 + cl];
#pragma unroll
      for (int i = 0; i < 8; ++i) {
        float v = silu_f(acc[nt][i] + bv);
        sh_a[wv][(i + 8 * hf) * 256 + nt * 16 + cl] = (__bf16)v;
      }
    }
    v8f acc2[NT8];
#pragma unroll
    for (int nt = 0; nt < NT8; ++nt) acc2[nt] = vz;
#pragma unroll
    for (int kt = 0; kt < 4; ++kt) {
      v16bf af = a_frag(sh_a[wv], 256, kt * 32);
#pragma unroll
      for (int nt = 0; nt < NT8; ++nt) {
        v16bf bf = b_frag(w2s, nt * 4 + kt);
        acc2[nt] = wmma_bf16(af, bf, acc2[nt]);
      }
    }
#pragma unroll
    for (int nt = 0; nt < NT8; ++nt) {
      float bv = hb2[nt * 16 + cl];
#pragma unroll
      for (int i = 0; i < 8; ++i) {
        int node = nbase + i + 8 * hf;
        if (node < n) {
          size_t idx = (size_t)node * DIM + nt * 16 + cl;
          hn[idx] = hn[idx] + acc2[nt][i] + bv;   // residual
        }
      }
    }
    if (lane < 16) {
      int node = nbase + lane;
      if (node < n) {
#pragma unroll
        for (int v = 0; v < 3; ++v) xn[node * 3 + v] += xagg[node * 3 + v];
      }
    }
  }
}

// ---------------------------------------------------------------------------
// Dense 128->128 MLP layer (relu) with WMMA (final MLP layers 0 and 1)
// ---------------------------------------------------------------------------
__global__ void __launch_bounds__(128) mlp_gemm_kernel(
    const float* __restrict__ src, float* __restrict__ dst,
    const __bf16* __restrict__ wp, const float* __restrict__ bias, int n) {
  __shared__ alignas(16) __bf16 ws_[4 * NT8 * 512];   // 32768 B
  __shared__ alignas(16) __bf16 sh_a[4][16 * 128];

  stage_lds(wp, ws_, 4 * NT8 * 512 * 2);
  stage_fence();

  const int lane = threadIdx.x & 31;
  const int wv = threadIdx.x >> 5;
  const int hf = lane >> 4, cl = lane & 15;
  const v8f vz = {0.f, 0.f, 0.f, 0.f, 0.f, 0.f, 0.f, 0.f};
  const int ntiles = (n + 15) / 16;

  for (int tile = blockIdx.x * 4 + wv; tile < ntiles; tile += gridDim.x * 4) {
    const int nbase = tile * 16;
    for (int r = 0; r < 16; ++r) {
      int node = nbase + r;
      if (node > n - 1) node = n - 1;
#pragma unroll
      for (int j = 0; j < 4; ++j) {
        int col = lane + 32 * j;
        sh_a[wv][r * 128 + col] = (__bf16)src[(size_t)node * DIM + col];
      }
    }
    v8f acc[NT8];
#pragma unroll
    for (int nt = 0; nt < NT8; ++nt) acc[nt] = vz;
#pragma unroll
    for (int kt = 0; kt < 4; ++kt) {
      v16bf af = a_frag(sh_a[wv], 128, kt * 32);
#pragma unroll
      for (int nt = 0; nt < NT8; ++nt) {
        v16bf bf = b_frag(ws_, nt * 4 + kt);
        acc[nt] = wmma_bf16(af, bf, acc[nt]);
      }
    }
#pragma unroll
    for (int nt = 0; nt < NT8; ++nt) {
      float bv = bias[nt * 16 + cl];
#pragma unroll
      for (int i = 0; i < 8; ++i) {
        int node = nbase + i + 8 * hf;
        if (node < n)
          dst[(size_t)node * DIM + nt * 16 + cl] = fmaxf(acc[nt][i] + bv, 0.0f);
      }
    }
  }
}

// ---------------------------------------------------------------------------
// Pooling score: tanh(h . w / ||w||)
// ---------------------------------------------------------------------------
__global__ void score_kernel(const float* __restrict__ hn,
                             const float* __restrict__ w,
                             float* __restrict__ score, int n) {
  int i = blockIdx.x * blockDim.x + threadIdx.x;
  if (i >= n) return;
  float nrm = 0.0f, s = 0.0f;
  for (int c = 0; c < DIM; ++c) {
    float wc = w[c];
    nrm += wc * wc;
    s += hn[(size_t)i * DIM + c] * wc;
  }
  score[i] = tanhf(s * __frsqrt_rn(nrm));
}

// Top-K (K=5) per graph, one block per graph.
__global__ void topk_kernel(const float* __restrict__ score, int* __restrict__ perm,
                            float* __restrict__ sc) {
  __shared__ float s[NPER];
  __shared__ float rv[256];
  __shared__ int ri[256];
  const int g = blockIdx.x, tid = threadIdx.x;
  for (int i = tid; i < NPER; i += 256) s[i] = score[g * NPER + i];
  __syncthreads();
  for (int kk = 0; kk < TOPK; ++kk) {
    float best = -1e30f;
    int bi = 0;
    for (int i = tid; i < NPER; i += 256) {
      float v = s[i];
      if (v > best) { best = v; bi = i; }
    }
    rv[tid] = best;
    ri[tid] = bi;
    __syncthreads();
    for (int off = 128; off > 0; off >>= 1) {
      if (tid < off) {
        if (rv[tid + off] > rv[tid] ||
            (rv[tid + off] == rv[tid] && ri[tid + off] < ri[tid])) {
          rv[tid] = rv[tid + off];
          ri[tid] = ri[tid + off];
        }
      }
      __syncthreads();
    }
    if (tid == 0) {
      perm[g * TOPK + kk] = g * NPER + ri[0];
      sc[g * TOPK + kk] = rv[0];
      s[ri[0]] = -1e30f;
    }
    __syncthreads();
  }
}

// Gather pooled features (scaled by score) + scatter new_id.
__global__ void pool_gather_kernel(const float* __restrict__ hn,
                                   const float* __restrict__ xn,
                                   const int* __restrict__ perm,
                                   const float* __restrict__ sc,
                                   float* __restrict__ Hp, float* __restrict__ Xp,
                                   int* __restrict__ new_id) {
  int t = blockIdx.x * blockDim.x + threadIdx.x;
  if (t < NPOOL * DIM) {
    int j = t >> 7, c = t & 127;
    Hp[t] = hn[(size_t)perm[j] * DIM + c] * sc[j];
  }
  if (t < NPOOL * 3) {
    int j = t / 3, v = t - j * 3;
    Xp[t] = xn[(size_t)perm[j] * 3 + v] * sc[j];
  }
  if (t < NPOOL) new_id[perm[t]] = t;
}

// knn interpolation (K == 5 pooled nodes of the node's own graph)
__global__ void knn_kernel(const float* __restrict__ xl, const float* __restrict__ hl,
                           const float* __restrict__ Xp, const float* __restrict__ Hp,
                           float* __restrict__ xb, float* __restrict__ hb,
                           float* __restrict__ Xx_out, int n) {
  int i = blockIdx.x * blockDim.x + threadIdx.x;
  if (i >= n) return;
  int g = i / NPER;
  float w[TOPK];
  float wsum = 0.0f;
#pragma unroll
  for (int k = 0; k < TOPK; ++k) {
    int p = g * TOPK + k;
    float sq = 0.0f;
    for (int v = 0; v < 3; ++v) {
      float d = Xp[p * 3 + v] - xl[i * 3 + v];
      sq += d * d;
    }
    for (int c = 0; c < DIM; ++c) {
      float d = Hp[(size_t)p * DIM + c] - hl[(size_t)i * DIM + c];
      sq += d * d;
    }
    float wk = 1.0f / fmaxf(sq, 1e-16f);
    w[k] = wk;
    wsum += wk;
  }
  float inv = 1.0f / wsum;
#pragma unroll
  for (int v = 0; v < 3; ++v) {
    float xx = 0.0f;
#pragma unroll
    for (int k = 0; k < TOPK; ++k) xx += w[k] * Xp[(g * TOPK + k) * 3 + v];
    xx *= inv;
    Xx_out[i * 3 + v] = xx;
    xb[i * 3 + v] = 0.5f * (xl[i * 3 + v] + xx);
  }
  for (int c = 0; c < DIM; ++c) {
    float hh = 0.0f;
#pragma unroll
    for (int k = 0; k < TOPK; ++k) hh += w[k] * Hp[(size_t)(g * TOPK + k) * DIM + c];
    hb[(size_t)i * DIM + c] = hl[(size_t)i * DIM + c] + 0.5f * hh * inv;
  }
}

// Final tiny layer: m = mid2 @ W2 + b2 ; out_x = (x_l - Xx)/2 * m + Xx
__global__ void final_kernel(const float* __restrict__ mid2,
                             const float* __restrict__ W2, const float* __restrict__ b2,
                             const float* __restrict__ xl, const float* __restrict__ Xx,
                             float* __restrict__ out, int n) {
  int i = blockIdx.x * blockDim.x + threadIdx.x;
  if (i >= n) return;
  float m0 = b2[0], m1 = b2[1], m2 = b2[2];
  for (int k = 0; k < DIM; ++k) {
    float t = mid2[(size_t)i * DIM + k];
    m0 += t * W2[k * 3 + 0];
    m1 += t * W2[k * 3 + 1];
    m2 += t * W2[k * 3 + 2];
  }
  float mm[3] = {m0, m1, m2};
#pragma unroll
  for (int v = 0; v < 3; ++v)
    out[i * 3 + v] = 0.5f * (xl[i * 3 + v] - Xx[i * 3 + v]) * mm[v] + Xx[i * 3 + v];
}

// ---------------------------------------------------------------------------
// Host orchestration
// ---------------------------------------------------------------------------
extern "C" void kernel_launch(void* const* d_in, const int* in_sizes, int n_in,
                              void* d_out, int out_size, void* d_ws, size_t ws_size,
                              hipStream_t stream) {
  (void)in_sizes; (void)n_in; (void)out_size; (void)ws_size;

  const float* eattr = (const float*)d_in[0];
  const float* x_l = (const float*)d_in[1];
  const float* h_l = (const float*)d_in[2];
  // params pytree (sorted dict keys): egnn_hidden, egnn_in, egnn_out, mlp, pool_w
  const int BASE_HID = 3, BASE_IN = 23, BASE_OUT = 43, BASE_MLP = 63, IDX_POOLW = 69;
  const int* esrc = (const int*)d_in[70];
  const int* edst = esrc + NEDGES;

  auto LW = [&](int base, int l, int j) -> const float* {
    // per-layer sorted leaves: e1.W,e1.b,e2.W,e2.b,h1.W,h1.b,h2.W,h2.b,xw.W,xw.b
    return (const float*)d_in[base + 10 * l + j];
  };

  char* ws = (char*)d_ws;
  size_t off = 0;
  auto alloc = [&](size_t bytes) -> char* {
    char* p = ws + off;
    off = (off + bytes + 255) & ~(size_t)255;
    return p;
  };

  struct PackSet { __bf16 *e1, *e2, *h1, *h2; };
  PackSet pk[6];
  for (int i = 0; i < 6; ++i) {
    pk[i].e1 = (__bf16*)alloc(36864 * 2);   // 8*9*512
    pk[i].e2 = (__bf16*)alloc(16384 * 2);   // 8*4*512
    pk[i].h1 = (__bf16*)alloc(32768 * 2);   // 8*8*512
    pk[i].h2 = (__bf16*)alloc(16384 * 2);
  }
  __bf16* mp0 = (__bf16*)alloc(16384 * 2);
  __bf16* mp1 = (__bf16*)alloc(16384 * 2);

  float* h_work = (float*)alloc((size_t)NNODES * DIM * 4);
  float* x_work = (float*)alloc((size_t)NNODES * 3 * 4);
  float* magg = (float*)alloc((size_t)NNODES * DIM * 4);
  float* xagg = (float*)alloc((size_t)NNODES * 3 * 4);
  float* score = (float*)alloc((size_t)NNODES * 4);
  int* perm = (int*)alloc(NPOOL * 4);
  float* sc = (float*)alloc(NPOOL * 4);
  int* new_id = (int*)alloc((size_t)NNODES * 4);
  float* Hp = (float*)alloc((size_t)NPOOL * DIM * 4);
  float* Xp = (float*)alloc((size_t)NPOOL * 3 * 4);
  float* Xx = (float*)alloc((size_t)NNODES * 3 * 4);
  float* h_b = (float*)alloc((size_t)NNODES * DIM * 4);
  float* x_b = (float*)alloc((size_t)NNODES * 3 * 4);
  float* mid1 = magg;    // reuse after egnn stages complete
  float* mid2 = h_work;  // reuse after pooling/knn complete

  // ---- pack all weights to WMMA bf16 fragment layout ----------------------
  for (int i = 0; i < 6; ++i) {
    int base = (i < 2) ? BASE_IN : (i < 4) ? BASE_HID : BASE_OUT;
    int l = i & 1;
    pack_w_kernel<<<144, 256, 0, stream>>>(LW(base, l, 0), pk[i].e1, E1K, DIM, E1KT, NT8);
    pack_w_kernel<<<64, 256, 0, stream>>>(LW(base, l, 2), pk[i].e2, DIM, DIM, 4, NT8);
    pack_w_kernel<<<128, 256, 0, stream>>>(LW(base, l, 4), pk[i].h1, 2 * DIM, DIM, 8, NT8);
    pack_w_kernel<<<64, 256, 0, stream>>>(LW(base, l, 6), pk[i].h2, DIM, DIM, 4, NT8);
  }
  pack_w_kernel<<<64, 256, 0, stream>>>((const float*)d_in[BASE_MLP + 0], mp0, DIM, DIM, 4, NT8);
  pack_w_kernel<<<64, 256, 0, stream>>>((const float*)d_in[BASE_MLP + 2], mp1, DIM, DIM, 4, NT8);

  (void)hipMemcpyAsync(h_work, h_l, (size_t)NNODES * DIM * 4,
                       hipMemcpyDeviceToDevice, stream);
  (void)hipMemcpyAsync(x_work, x_l, (size_t)NNODES * 3 * 4,
                       hipMemcpyDeviceToDevice, stream);

  auto gemm_blocks = [](int nn) {
    int ntiles = (nn + 15) / 16;
    int nb = (ntiles + 3) / 4;
    return nb > MAX_GEMM_BLOCKS ? MAX_GEMM_BLOCKS : nb;
  };

  auto run_egnn = [&](int i, int base, int l, float* xn, float* hn,
                      const int* nid, int nn) {
    (void)hipMemsetAsync(magg, 0, (size_t)nn * DIM * 4, stream);
    (void)hipMemsetAsync(xagg, 0, (size_t)nn * 3 * 4, stream);
    edge_kernel<<<EDGE_BLOCKS, 128, 0, stream>>>(
        esrc, edst, eattr, xn, hn, pk[i].e1, LW(base, l, 1), pk[i].e2,
        LW(base, l, 3), LW(base, l, 8), LW(base, l, 9), nid, magg, xagg);
    node_kernel<<<gemm_blocks(nn), 128, 0, stream>>>(
        xn, hn, magg, xagg, pk[i].h1, LW(base, l, 5), pk[i].h2, LW(base, l, 7), nn);
  };

  // ---- egnn_in ------------------------------------------------------------
  run_egnn(0, BASE_IN, 0, x_work, h_work, nullptr, NNODES);
  run_egnn(1, BASE_IN, 1, x_work, h_work, nullptr, NNODES);

  // ---- TopK pooling -------------------------------------------------------
  score_kernel<<<(NNODES + 255) / 256, 256, 0, stream>>>(
      h_work, (const float*)d_in[IDX_POOLW], score, NNODES);
  topk_kernel<<<NGRAPHS, 256, 0, stream>>>(score, perm, sc);
  (void)hipMemsetAsync(new_id, 0xFF, (size_t)NNODES * 4, stream);  // -1
  pool_gather_kernel<<<50, 256, 0, stream>>>(h_work, x_work, perm, sc, Hp, Xp, new_id);

  // ---- egnn_hidden on pooled graph (masked edges) -------------------------
  run_egnn(2, BASE_HID, 0, Xp, Hp, new_id, NPOOL);
  run_egnn(3, BASE_HID, 1, Xp, Hp, new_id, NPOOL);

  // ---- knn interpolation --------------------------------------------------
  knn_kernel<<<(NNODES + 255) / 256, 256, 0, stream>>>(x_l, h_l, Xp, Hp, x_b, h_b,
                                                       Xx, NNODES);

  // ---- egnn_out -----------------------------------------------------------
  run_egnn(4, BASE_OUT, 0, x_b, h_b, nullptr, NNODES);
  run_egnn(5, BASE_OUT, 1, x_b, h_b, nullptr, NNODES);

  // ---- final MLP + output -------------------------------------------------
  mlp_gemm_kernel<<<gemm_blocks(NNODES), 128, 0, stream>>>(
      h_b, mid1, mp0, (const float*)d_in[BASE_MLP + 1], NNODES);
  mlp_gemm_kernel<<<gemm_blocks(NNODES), 128, 0, stream>>>(
      mid1, mid2, mp1, (const float*)d_in[BASE_MLP + 3], NNODES);
  final_kernel<<<(NNODES + 255) / 256, 256, 0, stream>>>(
      mid2, (const float*)d_in[BASE_MLP + 4], (const float*)d_in[BASE_MLP + 5],
      x_l, Xx, (float*)d_out, NNODES);
  // second output: rh
  (void)hipMemcpyAsync((float*)d_out + (size_t)NNODES * 3, h_b,
                       (size_t)NNODES * DIM * 4, hipMemcpyDeviceToDevice, stream);
}
